// RoPEAttention_52596169507611
// MI455X (gfx1250) — matrix operand order repaired and checked by reading, not compiled, not run on hardware
//
#include <hip/hip_runtime.h>
#include <hip/hip_bf16.h>
#include <math.h>

#define DIM   1024
#define NH    16
#define HD    64
#define BATCH 4
#define SEQ   2048
#define MROWS (BATCH*SEQ)

typedef _Float16 v16h __attribute__((ext_vector_type(16)));
typedef float    v8f  __attribute__((ext_vector_type(8)));
typedef _Float16 h4   __attribute__((ext_vector_type(4)));
typedef float    f4   __attribute__((ext_vector_type(4)));

union FragH { v16h h; unsigned u[8]; };
union FragF { v8f  v; float    e[8]; };

static __device__ inline v8f wmma_f16(const FragH& a, const FragH& b, v8f c) {
  // D = A(16x32 f16) * B(32x16 f16) + C(16x16 f32)
  return __builtin_amdgcn_wmma_f32_16x16x32_f16(false, a.h, false, b.h,
                                                (short)0, c, false, false);
}

// ---------------------------------------------------------------------------
// Tiled GEMM: C[M=8192, N=1024] = A[8192,1024] * B[1024,1024]
// 128x128 block tile, K-step 64 (16 WMMAs per wave between barriers).
// AHALF: A is f16 (attention output) else f32 (x).
// EPI 0: write f32 row-major to outF (final projection -> d_out)
// EPI 1: RoPE in epilogue, write f16 [b,h,t,hd] to outH   (Q, K)
// EPI 2: write f16 [b,h,t,hd] to outH                     (V)
// ---------------------------------------------------------------------------
template<bool AHALF, int EPI>
__global__ __launch_bounds__(256) void gemm128(
    const void* __restrict__ Aptr, const float* __restrict__ Bw,
    float* __restrict__ outF, _Float16* __restrict__ outH,
    const float* __restrict__ cosp, const float* __restrict__ sinp)
{
  __shared__ alignas(16) _Float16 lA [128*64];   // [m][k] row-major (16 KB)
  __shared__ alignas(16) _Float16 lBt[128*64];   // [n][k] transposed (16 KB)

  const int tid  = threadIdx.x;
  const int lane = tid & 31, wid = tid >> 5;
  const int wm = wid >> 1, wn = wid & 1;         // 4 x 2 wave grid
  const int ln = lane & 15, khf = lane >> 4;
  const int blkM = blockIdx.y * 128, blkN = blockIdx.x * 128;

  FragF acc[2][4];
#pragma unroll
  for (int mf=0; mf<2; ++mf)
#pragma unroll
    for (int nf=0; nf<4; ++nf)
#pragma unroll
      for (int r=0; r<8; ++r) acc[mf][nf].e[r] = 0.f;

  for (int k0 = 0; k0 < DIM; k0 += 64) {
    __syncthreads();                              // previous compute done
    // ---- stage A (128x64) and B (64x128, transposed) into LDS as f16 ----
#pragma unroll
    for (int i=0;i<8;i++){
      int vidx = tid + i*256;                     // 0..2047 float4 groups
      int ar = vidx >> 4, ac = (vidx & 15) << 2;  // A: 16 groups per row
      h4 hv;
      if (AHALF) {
        hv = *(const h4*)((const _Float16*)Aptr + (size_t)(blkM + ar)*DIM + k0 + ac);
      } else {
        f4 fv = *(const f4*)((const float*)Aptr + (size_t)(blkM + ar)*DIM + k0 + ac);
        hv[0]=(_Float16)fv[0]; hv[1]=(_Float16)fv[1];
        hv[2]=(_Float16)fv[2]; hv[3]=(_Float16)fv[3];
      }
      *(h4*)&lA[ar*64 + ac] = hv;

      int br = vidx >> 5, bn = (vidx & 31) << 2;  // B: 32 groups per k-row, 64 rows
      f4 wv = *(const f4*)(Bw + (size_t)(k0 + br)*DIM + blkN + bn);
      lBt[(bn+0)*64 + br] = (_Float16)wv[0];
      lBt[(bn+1)*64 + br] = (_Float16)wv[1];
      lBt[(bn+2)*64 + br] = (_Float16)wv[2];
      lBt[(bn+3)*64 + br] = (_Float16)wv[3];
    }
    if (k0 + 64 < DIM) {                          // prefetch next K tile
      __builtin_prefetch((const char*)Aptr +
          ((size_t)(blkM + (tid & 127))*DIM + k0 + 64) * (AHALF ? 2 : 4), 0, 3);
      __builtin_prefetch((const char*)(Bw + (size_t)(k0 + 64 + (tid & 63))*DIM + blkN), 0, 3);
    }
    __syncthreads();

    // ---- fragment loads (ISA VGPR layouts), two 32-wide K chunks ----
    FragH a[2][2], b[4][2];
#pragma unroll
    for (int mf=0; mf<2; ++mf){
      int m = wm*32 + mf*16 + ln;
#pragma unroll
      for (int c=0; c<2; ++c)
#pragma unroll
        for (int i=0;i<8;i++){
          int kk = c*32 + ((i>>2)<<4) + (khf<<3) + ((i&3)<<1);   // A layout
          a[mf][c].u[i] = *(const unsigned*)&lA[m*64 + kk];
        }
    }
#pragma unroll
    for (int nf=0; nf<4; ++nf){
      int n = wn*64 + nf*16 + ln;
#pragma unroll
      for (int c=0; c<2; ++c)
#pragma unroll
        for (int i=0;i<8;i++){
          int kk = c*32 + ((i>>2)<<4) + (khf<<3) + ((i&3)<<1);   // B layout
          b[nf][c].u[i] = *(const unsigned*)&lBt[n*64 + kk];
        }
    }
#pragma unroll
    for (int mf=0; mf<2; ++mf)
#pragma unroll
      for (int nf=0; nf<4; ++nf){
        acc[mf][nf].v = wmma_f16(a[mf][0], b[nf][0], acc[mf][nf].v);
        acc[mf][nf].v = wmma_f16(a[mf][1], b[nf][1], acc[mf][nf].v);
      }
  }

  // ---- epilogue ----
#pragma unroll
  for (int mf=0; mf<2; ++mf){
#pragma unroll
    for (int nf=0; nf<4; ++nf){
#pragma unroll
      for (int r=0; r<8; ++r){
        int row = blkM + wm*32 + mf*16 + ((lane < 16) ? r : r + 8);
        int col = blkN + wn*64 + nf*16 + ln;
        float val = acc[mf][nf].e[r];
        if (EPI == 0) {
          outF[(size_t)row*DIM + col] = val;
        } else {
          int t  = row & (SEQ-1), bb = row >> 11;  // SEQ = 2048 = 2^11
          int hh = col >> 6,      d  = col & 63;
          float res = val;
          if (EPI == 1) {
            // rotation pair (d even, d+1) sits in adjacent lanes
            float other = __shfl_xor(val, 1, 32);
            int pi = d >> 1;
            float c = cosp[t*(HD/2) + pi];
            float s = sinp[t*(HD/2) + pi];
            res = ((d & 1) == 0) ? (val*c - other*s) : (other*s + val*c);
          }
          outH[(((size_t)(bb*NH + hh))*SEQ + t)*HD + d] = (_Float16)res;
        }
      }
    }
  }
}

// ---------------------------------------------------------------------------
// Flash attention: block = (b, h, 64-query tile), 4 waves x 16 query rows.
// q/k/v layout: f16 [b][h][t][64]. Output: f16 [b][t][DIM].
// K tile staged with GLOBAL_LOAD_ASYNC_TO_LDS_B128 (ASYNCcnt tracked);
// V tile staged synchronously because its LDS copy is transposed.
// ---------------------------------------------------------------------------
__global__ __launch_bounds__(128) void attn_fa(
    const _Float16* __restrict__ q, const _Float16* __restrict__ k,
    const _Float16* __restrict__ vv, _Float16* __restrict__ out)
{
  __shared__ alignas(16) _Float16 lK [64*64];      // [key][d]
  __shared__ alignas(16) _Float16 lVt[64*64];      // [d][key] (key contiguous)
  __shared__ alignas(16) _Float16 lP [4][16*64];   // per-wave P transpose buffer

  const int tid = threadIdx.x, lane = tid & 31, wid = tid >> 5;
  const int ln = lane & 15, khf = lane >> 4;
  const int qt = blockIdx.x, hh = blockIdx.y, bb = blockIdx.z;
  const size_t bhOff = ((size_t)(bb*NH + hh)) * SEQ * HD;
  const _Float16* qbh = q  + bhOff;
  const _Float16* kbh = k  + bhOff;
  const _Float16* vbh = vv + bhOff;

  // Q A-fragments (16 rows x 64 d, two 16x32 chunks), loaded once
  FragH aQ[2];
  {
    int tq = qt*64 + wid*16 + ln;
#pragma unroll
    for (int c=0;c<2;c++)
#pragma unroll
      for (int i=0;i<8;i++){
        int d = c*32 + ((i>>2)<<4) + (khf<<3) + ((i&3)<<1);
        aQ[c].u[i] = *(const unsigned*)(qbh + (size_t)tq*HD + d);
      }
  }

  FragF o[4];
  float mr[8], lr[8];
#pragma unroll
  for (int nf=0;nf<4;nf++)
#pragma unroll
    for (int r=0;r<8;r++) o[nf].e[r]=0.f;
#pragma unroll
  for (int r=0;r<8;r++){ mr[r] = -1e30f; lr[r]=0.f; }

  for (int j=0;j<SEQ/64;j++){
    __syncthreads();                               // prev tile's LDS reads done

    // ---- K tile: async global -> LDS (no VGPR round-trip, ASYNCcnt) ----
    {
      unsigned long long kbase = (unsigned long long)(kbh + (size_t)j*64*HD);
#pragma unroll
      for (int i=0;i<8;i++){
        int vi = tid + i*128;                      // 0..1023 h4 groups
        int key = vi >> 4, c4 = (vi & 15) << 2;
        unsigned ldsoff = (unsigned)(uintptr_t)(&lK[key*64 + c4]);
        unsigned goff   = (unsigned)((key*64 + c4) * 2);
        asm volatile("global_load_async_to_lds_b128 %0, %1, %2"
                     :: "v"(ldsoff), "v"(goff), "s"(kbase) : "memory");
      }
    }
    // ---- V tile: synchronous transpose into LDS ----
#pragma unroll
    for (int i=0;i<8;i++){
      int vi = tid + i*128;
      int key = vi >> 4, c4 = (vi & 15) << 2;
      h4 vvec = *(const h4*)(vbh + (size_t)(j*64+key)*HD + c4);
      lVt[(c4+0)*64 + key] = vvec[0];
      lVt[(c4+1)*64 + key] = vvec[1];
      lVt[(c4+2)*64 + key] = vvec[2];
      lVt[(c4+3)*64 + key] = vvec[3];
    }
    asm volatile("s_wait_asynccnt 0x0" ::: "memory");   // K tile landed in LDS
    __syncthreads();

    // ---- S = Q * K^T  (scale 1/sqrt(64) = 0.125) ----
    FragF s[4];
#pragma unroll
    for (int nf=0;nf<4;nf++){
      FragH b0, b1;
      int key = nf*16 + ln;
#pragma unroll
      for (int i=0;i<8;i++){
        int d0 = ((i>>2)<<4) + (khf<<3) + ((i&3)<<1);
        b0.u[i] = *(const unsigned*)&lK[key*64 + d0];
        b1.u[i] = *(const unsigned*)&lK[key*64 + 32 + d0];
      }
      v8f z = {};
      z = wmma_f16(aQ[0], b0, z);
      s[nf].v = wmma_f16(aQ[1], b1, z);
#pragma unroll
      for (int r=0;r<8;r++) s[nf].e[r] *= 0.125f;
    }

    // ---- online softmax (row reductions across 16-lane halves) ----
    float mn[8], al[8], rs[8];
#pragma unroll
    for (int r=0;r<8;r++){
      float x = fmaxf(fmaxf(s[0].e[r], s[1].e[r]), fmaxf(s[2].e[r], s[3].e[r]));
      x = fmaxf(x, __shfl_xor(x, 1, 32));
      x = fmaxf(x, __shfl_xor(x, 2, 32));
      x = fmaxf(x, __shfl_xor(x, 4, 32));
      x = fmaxf(x, __shfl_xor(x, 8, 32));
      mn[r] = fmaxf(mr[r], x);
      al[r] = __expf(mr[r] - mn[r]);
      mr[r] = mn[r];
      rs[r] = 0.f;
    }
#pragma unroll
    for (int nf=0;nf<4;nf++)
#pragma unroll
      for (int r=0;r<8;r++){
        float p = __expf(s[nf].e[r] - mn[r]);
        s[nf].e[r] = p;
        rs[r] += p;
      }
#pragma unroll
    for (int r=0;r<8;r++){
      rs[r] += __shfl_xor(rs[r], 1, 32);
      rs[r] += __shfl_xor(rs[r], 2, 32);
      rs[r] += __shfl_xor(rs[r], 4, 32);
      rs[r] += __shfl_xor(rs[r], 8, 32);
      lr[r] = lr[r]*al[r] + rs[r];
    }
#pragma unroll
    for (int nf=0;nf<4;nf++)
#pragma unroll
      for (int r=0;r<8;r++) o[nf].e[r] *= al[r];

    // ---- P: C-layout -> A-layout via per-wave LDS transpose ----
#pragma unroll
    for (int nf=0;nf<4;nf++)
#pragma unroll
      for (int r=0;r<8;r++){
        int qr = (lane<16) ? r : r+8;
        lP[wid][qr*64 + nf*16 + ln] = (_Float16)s[nf].e[r];
      }
    __syncthreads();

    FragH aP[2];
#pragma unroll
    for (int c=0;c<2;c++)
#pragma unroll
      for (int i=0;i<8;i++){
        int key = c*32 + ((i>>2)<<4) + (khf<<3) + ((i&3)<<1);
        aP[c].u[i] = *(const unsigned*)&lP[wid][ln*64 + key];
      }

    // ---- O += P * V ----
#pragma unroll
    for (int nf=0;nf<4;nf++){
      FragH b0,b1;
      int nd = nf*16 + ln;
#pragma unroll
      for (int i=0;i<8;i++){
        int kk = ((i>>2)<<4) + (khf<<3) + ((i&3)<<1);
        b0.u[i] = *(const unsigned*)&lVt[nd*64 + kk];
        b1.u[i] = *(const unsigned*)&lVt[nd*64 + 32 + kk];
      }
      o[nf].v = wmma_f16(aP[0], b0, o[nf].v);
      o[nf].v = wmma_f16(aP[1], b1, o[nf].v);
    }
  }

  // ---- normalize and write f16 [b][t][DIM] ----
#pragma unroll
  for (int nf=0;nf<4;nf++)
#pragma unroll
    for (int r=0;r<8;r++){
      int qr = (lane<16) ? r : r+8;
      int t = qt*64 + wid*16 + qr;
      float val = o[nf].e[r] / lr[r];
      out[((size_t)bb*SEQ + t)*DIM + hh*64 + nf*16 + ln] = (_Float16)val;
    }
}

// ---------------------------------------------------------------------------
extern "C" void kernel_launch(void* const* d_in, const int* in_sizes, int n_in,
                              void* d_out, int out_size, void* d_ws, size_t ws_size,
                              hipStream_t stream) {
  const float* x  = (const float*)d_in[0];
  const float* wq = (const float*)d_in[1];
  const float* wk = (const float*)d_in[2];
  const float* wv = (const float*)d_in[3];
  const float* wo = (const float*)d_in[4];
  const float* fc = (const float*)d_in[5];
  const float* fs = (const float*)d_in[6];

  _Float16* qh = (_Float16*)d_ws;                    // [b,h,t,hd] f16, 16 MB each
  _Float16* kh = qh + (size_t)MROWS*DIM;
  _Float16* vh = kh + (size_t)MROWS*DIM;
  _Float16* at = vh + (size_t)MROWS*DIM;             // [b,t,DIM] f16

  dim3 g(DIM/128, MROWS/128), blk(256);
  gemm128<false,1><<<g, blk, 0, stream>>>(x, wq, nullptr, qh, fc, fs);
  gemm128<false,1><<<g, blk, 0, stream>>>(x, wk, nullptr, kh, fc, fs);
  gemm128<false,2><<<g, blk, 0, stream>>>(x, wv, nullptr, vh, nullptr, nullptr);

  attn_fa<<<dim3(SEQ/64, NH, BATCH), 128, 0, stream>>>(qh, kh, vh, at);

  gemm128<true,0><<<g, blk, 0, stream>>>(at, wo, (float*)d_out, nullptr, nullptr, nullptr);
}
